// Quantize_52475910422692
// MI455X (gfx1250) — matrix operand order repaired and checked by reading, not compiled, not run on hardware
//
#include <hip/hip_runtime.h>
#include <hip/hip_bf16.h>

typedef __bf16 bf16_t;
typedef __attribute__((ext_vector_type(16))) __bf16 v16bf;
typedef __attribute__((ext_vector_type(8)))  __bf16 v8bf;
typedef __attribute__((ext_vector_type(8)))  float  v8f;
typedef __attribute__((ext_vector_type(4)))  float  v4f;
typedef __attribute__((ext_vector_type(4)))  unsigned int u32x4;
typedef __attribute__((ext_vector_type(8)))  unsigned int u32x8;

#define DECAYF       0.99f
#define ONE_M_DECAY  0.01f
#define EPSF         1e-5f

constexpr int Nrows  = 16384;   // B*H*W
constexpr int Ddim   = 256;
constexpr int Kcodes = 8192;
constexpr int MBLK   = 64;      // rows per workgroup in argmin kernel
constexpr int CTile  = 64;      // codes per LDS tile iteration
constexpr int ASTR   = 264;     // padded LDS row stride (bf16) for A tile
constexpr int BSTR   = 72;      // padded LDS row stride (bf16) for B tile (TDM pad)
constexpr int BUFB   = 256 * BSTR * 2;   // 36,864 B per B tile buffer
constexpr int NIT    = Kcodes / CTile;   // 128

// ---------------------------------------------------------------------------
// K1: ||e_k||^2, bf16 copy of embed (D,K), and new_cluster_size = DECAY*cs
// ---------------------------------------------------------------------------
__global__ __launch_bounds__(256) void vq_prep_kernel(
    const float* __restrict__ embed, const float* __restrict__ cluster_size,
    bf16_t* __restrict__ embB, float* __restrict__ ebn, float* __restrict__ ncs_out)
{
  int k = blockIdx.x * 256 + threadIdx.x;
  float acc = 0.f;
  for (int d = 0; d < Ddim; ++d) {
    float v = embed[(size_t)d * Kcodes + k];
    acc += v * v;
    embB[(size_t)d * Kcodes + k] = (bf16_t)v;
  }
  ebn[k] = acc;
  ncs_out[k] = cluster_size[k] * DECAYF;
}

// ---------------------------------------------------------------------------
// K1b: new_embed_avg initialized to DECAY * embed_avg (atomics add the rest)
// ---------------------------------------------------------------------------
__global__ __launch_bounds__(256) void vq_eavg_init_kernel(
    const float* __restrict__ embed_avg, float* __restrict__ eavg_out, int n)
{
  int i = blockIdx.x * 256 + threadIdx.x;
  if (i < n) eavg_out[i] = embed_avg[i] * DECAYF;
}

// ---------------------------------------------------------------------------
// TDM: issue one 2D tensor-tile DMA  embB[0:256][c0:c0+64] -> LDS (padded rows)
//   data_size=2B; tile 64 x 256; tensor dim0=8192 (stride 8192), dim1=256
//   pad: every 128B (pad_interval=4 => 2^(4+3) bytes) insert 16B (pad_amount=3)
//   => LDS row stride 144B == BSTR bf16 elements
// ---------------------------------------------------------------------------
__device__ __forceinline__ void tdm_issue_tile(const bf16_t* gbase, int c0,
                                               unsigned lds_byte_addr)
{
  unsigned long long ga = (unsigned long long)(uintptr_t)gbase
                        + (unsigned long long)c0 * 2ull;
  u32x4 g0;
  g0[0] = 1u;                                        // count=1, load, user D#
  g0[1] = lds_byte_addr;                             // lds_addr
  g0[2] = (unsigned)(ga & 0xFFFFFFFFu);              // global_addr[31:0]
  g0[3] = (unsigned)((ga >> 32) & 0x01FFFFFFu)       // global_addr[56:32]
        | (2u << 30);                                // type = 2 ("image")
  u32x8 g1;
  g1[0] = (1u << 16)                                 // data_size = 2 bytes
        | (1u << 20)                                 // pad_enable
        | (4u << 22)                                 // pad_interval: 2^(4+3)=128B
        | (3u << 25);                                // pad_amount: 4 dwords =16B
  g1[1] = ((unsigned)Kcodes & 0xFFFFu) << 16;        // tensor_dim0[15:0] @bit48
  g1[2] = ((unsigned)Kcodes >> 16)                   // tensor_dim0[31:16]
        | (((unsigned)Ddim & 0xFFFFu) << 16);        // tensor_dim1[15:0] @bit80
  g1[3] = ((unsigned)Ddim >> 16)
        | ((unsigned)CTile << 16);                   // tile_dim0 @bit112
  g1[4] = (unsigned)Ddim;                            // tile_dim1 @bit128
  g1[5] = (unsigned)Kcodes;                          // tensor_dim0_stride @bit160
  g1[6] = 0u;
  g1[7] = 0u;
  asm volatile("tensor_load_to_lds %0, %1" :: "s"(g0), "s"(g1) : "memory");
}

// ---------------------------------------------------------------------------
// K2: fused distance-GEMM + argmin.  score = ||e||^2 + (-2x)·e  via WMMA bf16
//     B tiles double-buffered in LDS via Tensor Data Mover.
// ---------------------------------------------------------------------------
__global__ __launch_bounds__(256) void vq_argmin_kernel(
    const float* __restrict__ x, const bf16_t* __restrict__ embB,
    const float* __restrict__ ebn, int* __restrict__ idx_out)
{
  __shared__ __align__(16) char smem[2 * BUFB];   // 73,728 B (A tile fits buf1)

  const int tid  = threadIdx.x;
  const int lane = tid & 31;
  const int wave = tid >> 5;
  const int wm   = wave >> 1;        // row group 0..3   (16 rows each)
  const int wc   = wave & 1;         // code half 0..1   (32 codes each)
  const int lrow = lane & 15;
  const int lhs  = lane >> 4;        // half-selector within wave
  const int n0   = blockIdx.x * MBLK;

  const unsigned lds_base = (unsigned)(uintptr_t)(&smem[0]);

  // kick off DMA of B tile 0 into buf0 while we stage A into buf1
  if (wave == 0) tdm_issue_tile(embB, 0, lds_base);

  // ---- phase 1: stage -2*x (bf16) for our 64 rows into buf1 ----
  bf16_t* ldsA = (bf16_t*)(smem + BUFB);
  struct alignas(8) bf4 { bf16_t v[4]; };
  #pragma unroll
  for (int p = 0; p < 16; ++p) {
    int row = p * 4 + (tid >> 6);
    int col = (tid & 63) * 4;
    v4f v = *(const v4f*)(x + (size_t)(n0 + row) * Ddim + col);
    bf4 t;
    t.v[0] = (bf16_t)(-2.0f * v.x);
    t.v[1] = (bf16_t)(-2.0f * v.y);
    t.v[2] = (bf16_t)(-2.0f * v.z);
    t.v[3] = (bf16_t)(-2.0f * v.w);
    *(bf4*)(ldsA + row * ASTR + col) = t;
  }
  __syncthreads();

  // ---- A fragments: 16 rows x 256(K) bf16 held in registers all loop ----
  v16bf afrag[8];
  {
    const bf16_t* Ar = ldsA + (wm * 16 + lrow) * ASTR;
    #pragma unroll
    for (int s = 0; s < 8; ++s) {
      v8bf lo = *(const v8bf*)(Ar + s * 32 + lhs * 8);
      v8bf hi = *(const v8bf*)(Ar + s * 32 + lhs * 8 + 16);
      afrag[s] = __builtin_shufflevector(lo, hi,
                   0,1,2,3,4,5,6,7,8,9,10,11,12,13,14,15);
    }
  }
  __syncthreads();   // buf1 free for TDM after this point

  float rmin[8];
  int   ridx[8];
  #pragma unroll
  for (int j = 0; j < 8; ++j) { rmin[j] = 3.4e38f; ridx[j] = 0; }

  for (int it = 0; it < NIT; ++it) {
    const int c0 = it * CTile;

    if (wave == 0) __builtin_amdgcn_s_wait_tensorcnt(0);  // buf[it&1] ready
    __syncthreads();
    if (wave == 0 && it + 1 < NIT)                        // DMA next tile while
      tdm_issue_tile(embB, c0 + CTile,                    // we compute this one
                     lds_base + (unsigned)(((it + 1) & 1) * BUFB));

    const bf16_t* ldsB = (const bf16_t*)(smem + (it & 1) * BUFB);

    // ---- each wave: 16 rows x 32 codes (2 WMMA tiles x 8 k-steps) ----
    #pragma unroll
    for (int t = 0; t < 2; ++t) {
      const int cbase = c0 + wc * 32 + t * 16;
      const float e2  = ebn[cbase + lrow];     // per-lane output-column const
      v8f c;
      #pragma unroll
      for (int j = 0; j < 8; ++j) c[j] = e2;   // C preloaded with ||e||^2
      #pragma unroll
      for (int s = 0; s < 8; ++s) {
        // B layout: lane <-> code column (lrow), elements walk k in table order
        const bf16_t* Br = ldsB + (s * 32 + lhs * 8) * BSTR + (wc * 32 + t * 16 + lrow);
        v16bf bfrag;
        #pragma unroll
        for (int e = 0; e < 8; ++e) { bfrag[e] = Br[e * BSTR]; bfrag[e + 8] = Br[(e + 16) * BSTR]; }
        c = __builtin_amdgcn_wmma_f32_16x16x32_bf16(
              false, afrag[s], false, bfrag, (short)0, c, false, false);
      }
      const int code = cbase + lrow;
      #pragma unroll
      for (int j = 0; j < 8; ++j) {
        if (c[j] < rmin[j]) { rmin[j] = c[j]; ridx[j] = code; }
      }
    }
    __syncthreads();   // all reads of buf[it&1] done before it becomes DMA dst
  }

  // ---- cross-lane / cross-wave argmin reduction through LDS ----
  float* redv = (float*)smem;                 // 64 x 32 floats (8 KB)
  int*   redi = (int*)(smem + MBLK * 32 * 4); // 64 x 32 ints   (8 KB)
  const int slot = wc * 16 + lrow;
  #pragma unroll
  for (int j = 0; j < 8; ++j) {
    int rl = wm * 16 + lhs * 8 + j;           // C row M = j + 8*(lane>=16)
    redv[rl * 32 + slot] = rmin[j];
    redi[rl * 32 + slot] = ridx[j];
  }
  __syncthreads();
  if (tid < MBLK) {
    float best = redv[tid * 32];
    int   bi   = redi[tid * 32];
    for (int s2 = 1; s2 < 32; ++s2) {
      float v = redv[tid * 32 + s2];
      if (v < best) { best = v; bi = redi[tid * 32 + s2]; }
    }
    idx_out[n0 + tid] = bi;
  }
}

// ---------------------------------------------------------------------------
// K3: quantize gather + MSE partials + EMA scatter (f32 global atomics)
// ---------------------------------------------------------------------------
__global__ __launch_bounds__(256) void vq_output_kernel(
    const float* __restrict__ x, const float* __restrict__ embed,
    const int* __restrict__ idx, float* __restrict__ q_out,
    float* __restrict__ eavg_out, float* __restrict__ ncs_out,
    float* __restrict__ loss_acc)
{
  __shared__ float red[256];
  const int tid = threadIdx.x;
  const int n0  = blockIdx.x * 64;
  float ls = 0.f;
  for (int rr = 0; rr < 64; ++rr) {
    int n = n0 + rr;
    int k = idx[n];
    float e  = embed[(size_t)tid * Kcodes + k];   // L2-resident column gather
    float xv = x[(size_t)n * Ddim + tid];
    q_out[(size_t)n * Ddim + tid] = e;            // quantize_st == quantize
    float df = e - xv;
    ls += df * df;
    atomicAdd(&eavg_out[(size_t)tid * Kcodes + k], ONE_M_DECAY * xv);
    if (tid == 0) atomicAdd(&ncs_out[k], ONE_M_DECAY);
  }
  red[tid] = ls;
  __syncthreads();
  for (int s = 128; s > 0; s >>= 1) {
    if (tid < s) red[tid] += red[tid + s];
    __syncthreads();
  }
  if (tid == 0) atomicAdd(loss_acc, red[0]);
}

// ---------------------------------------------------------------------------
// K4: n = sum(new_cluster_size); finalize latent_loss
// ---------------------------------------------------------------------------
__global__ __launch_bounds__(256) void vq_stats_kernel(
    const float* __restrict__ ncs, const float* __restrict__ loss_acc,
    float* __restrict__ nsum_out, float* __restrict__ loss_out)
{
  __shared__ float red[256];
  const int tid = threadIdx.x;
  float s = 0.f;
  for (int k = tid; k < Kcodes; k += 256) s += ncs[k];
  red[tid] = s;
  __syncthreads();
  for (int st = 128; st > 0; st >>= 1) {
    if (tid < st) red[tid] += red[tid + st];
    __syncthreads();
  }
  if (tid == 0) {
    nsum_out[0] = red[0];
    loss_out[0] = loss_acc[0] / (float)(Nrows * Ddim);
  }
}

// ---------------------------------------------------------------------------
// K5: new_embed = new_embed_avg / smoothed cluster size
// ---------------------------------------------------------------------------
__global__ __launch_bounds__(256) void vq_embed_kernel(
    const float* __restrict__ eavg, const float* __restrict__ ncs,
    const float* __restrict__ nsum, float* __restrict__ embed_out, int n)
{
  int i = blockIdx.x * 256 + threadIdx.x;
  if (i >= n) return;
  int k = i & (Kcodes - 1);
  float nv = nsum[0];
  float cs = (ncs[k] + EPSF) / (nv + (float)Kcodes * EPSF) * nv;
  embed_out[i] = eavg[i] / cs;
}

// ---------------------------------------------------------------------------
extern "C" void kernel_launch(void* const* d_in, const int* in_sizes, int n_in,
                              void* d_out, int out_size, void* d_ws, size_t ws_size,
                              hipStream_t stream)
{
  const float* x          = (const float*)d_in[0];  // (16,32,32,256)
  const float* embed      = (const float*)d_in[1];  // (256,8192)
  const float* cluster_sz = (const float*)d_in[2];  // (8192,)
  const float* embed_avg  = (const float*)d_in[3];  // (256,8192)

  float* out      = (float*)d_out;
  float* out_q    = out;                 // 4,194,304  quantize_st
  float* out_loss = out + 4194304;       // 1          latent_loss
  float* out_emb  = out + 4194305;       // 2,097,152  new_embed
  float* out_ncs  = out + 6291457;       // 8,192      new_cluster_size
  float* out_eavg = out + 6299649;       // 2,097,152  new_embed_avg

  char* ws        = (char*)d_ws;
  bf16_t* embB    = (bf16_t*)ws;               // 4,194,304 B  bf16 embed copy
  float*  ebn     = (float*)(ws + 4194304);    //    32,768 B  ||e_k||^2
  int*    idx     = (int*)(ws + 4227072);      //    65,536 B  argmin indices
  float*  loss_acc= (float*)(ws + 4292608);    // scalar accum
  float*  nsum    = (float*)(ws + 4292616);    // scalar n

  hipMemsetAsync(ws + 4292608, 0, 16, stream);

  vq_prep_kernel<<<Kcodes / 256, 256, 0, stream>>>(embed, cluster_sz, embB, ebn, out_ncs);
  vq_eavg_init_kernel<<<(Ddim * Kcodes) / 256, 256, 0, stream>>>(embed_avg, out_eavg, Ddim * Kcodes);
  vq_argmin_kernel<<<Nrows / MBLK, 256, 0, stream>>>(x, embB, ebn, idx);
  vq_output_kernel<<<Nrows / 64, 256, 0, stream>>>(x, embed, idx, out_q, out_eavg, out_ncs, loss_acc);
  vq_stats_kernel<<<1, 256, 0, stream>>>(out_ncs, loss_acc, nsum, out_loss);
  vq_embed_kernel<<<(Ddim * Kcodes) / 256, 256, 0, stream>>>(out_eavg, out_ncs, nsum, out_emb, Ddim * Kcodes);
}
